// BasicBlock_51719996178696
// MI455X (gfx1250) — 8-bit matrix operand order repaired and checked by reading, compiled, not run on hardware
//
#include <hip/hip_runtime.h>
#include <hip/hip_bf16.h>
#include <stdint.h>

#define EPSV 1e-5f

typedef __attribute__((ext_vector_type(4))) int v4i;
typedef __attribute__((ext_vector_type(8))) int v8i;

#define NB  32
#define CC  256
#define HH  56
#define WWD 56
#define HWS (HH*WWD)
#define PP  (NB*HWS)
#define K3TOT 2304

static __device__ __forceinline__ signed char sgn8(float v) {
    return v > 0.f ? (signed char)1 : (v < 0.f ? (signed char)-1 : (signed char)0);
}

static __device__ __forceinline__ v8i cat8(v4i a, v4i b) {
    v8i r;
    r[0]=a[0]; r[1]=a[1]; r[2]=a[2]; r[3]=a[3];
    r[4]=b[0]; r[5]=b[1]; r[6]=b[2]; r[7]=b[3];
    return r;
}

__global__ void k_scales(const float* __restrict__ w3, const float* __restrict__ wp,
                         float* __restrict__ s1, float* __restrict__ s2) {
    int co = threadIdx.x;
    float a = 0.f;
    for (int i = 0; i < K3TOT; ++i) a += fabsf(w3[(size_t)co * K3TOT + i]);
    s1[co] = a / (float)K3TOT;
    a = 0.f;
    for (int i = 0; i < CC; ++i) a += fabsf(wp[(size_t)co * CC + i]);
    s2[co] = a / (float)CC;
}

__global__ void k_zero(int* __restrict__ z) {
    z[threadIdx.x] = 0;
}

__global__ void k_pack3(const float* __restrict__ w3, signed char* __restrict__ ap) {
    int idx = blockIdx.x * blockDim.x + threadIdx.x;
    if (idx >= 9 * 4 * 16 * 1024) return;
    int byte = idx & 3, vg = (idx >> 2) & 7, lane = (idx >> 5) & 31;
    int mo = (idx >> 10) & 15, ko = (idx >> 14) & 3, tap = idx >> 16;
    int half = lane >> 4;
    int k  = ((vg >> 2) * 32) + half * 16 + ((vg & 3) * 4) + byte;
    int co = mo * 16 + (lane & 15);
    int ci = ko * 64 + k;
    ap[idx] = sgn8(w3[((size_t)co * CC + ci) * 9 + tap]);
}

__global__ void k_pack1(const float* __restrict__ wp, signed char* __restrict__ ap) {
    int idx = blockIdx.x * blockDim.x + threadIdx.x;
    if (idx >= 4 * 16 * 1024) return;
    int byte = idx & 3, vg = (idx >> 2) & 7, lane = (idx >> 5) & 31;
    int mo = (idx >> 10) & 15, ko = (idx >> 14) & 3;
    int half = lane >> 4;
    int k  = ((vg >> 2) * 32) + half * 16 + ((vg & 3) * 4) + byte;
    int co = mo * 16 + (lane & 15);
    int ci = ko * 64 + k;
    ap[idx] = sgn8(wp[(size_t)co * CC + ci]);
}

__global__ void k_sign1(const float* __restrict__ x, const float* __restrict__ b11,
                        signed char* __restrict__ sb) {
    size_t e = (size_t)blockIdx.x * blockDim.x + threadIdx.x;
    int c = (int)(e & 255);
    int p = (int)(e >> 8);
    int n = p / HWS, hw = p - n * HWS;
    sb[e] = sgn8(x[((size_t)n * CC + c) * HWS + hw] + b11[c]);
}

__global__ void k_gemm3(const signed char* __restrict__ sb,
                        const signed char* __restrict__ zb,
                        const signed char* __restrict__ ap,
                        const float* __restrict__ scale,
                        float* __restrict__ out) {
    int wave = threadIdx.x >> 5;
    int lane = threadIdx.x & 31;
    int half = lane >> 4;
    int moBase = (wave & 3) * 4;
    int po = (blockIdx.x << 1) + (wave >> 2);
    int p  = (po << 4) + (lane & 15);
    int n  = p / HWS, hw = p - n * HWS;
    int h  = hw / WWD, w = hw - h * WWD;

    v8i acc0 = {0,0,0,0,0,0,0,0};
    v8i acc1 = {0,0,0,0,0,0,0,0};
    v8i acc2 = {0,0,0,0,0,0,0,0};
    v8i acc3 = {0,0,0,0,0,0,0,0};

    for (int tap = 0; tap < 9; ++tap) {
        int dh = tap / 3 - 1;
        int dw = tap - (tap / 3) * 3 - 1;
        int h2 = h + dh, w2 = w + dw;
        bool valid = (h2 >= 0) & (h2 < HH) & (w2 >= 0) & (w2 < WWD);
        const signed char* bbase =
            valid ? sb + (((size_t)(n * HWS + h2 * WWD + w2)) << 8) : zb;
        #pragma unroll
        for (int ko = 0; ko < 4; ++ko) {
            const signed char* bb = bbase + ko * 64 + half * 16;
            v4i b0 = *(const v4i*)bb;
            v4i b1 = *(const v4i*)(bb + 32);
            v8i b = cat8(b0, b1);
            const signed char* abase = ap + ((((size_t)(tap * 4 + ko)) * 16 + moBase) << 10)
                                          + (lane << 5);
            v8i a0v = *(const v8i*)(abase);
            v8i a1v = *(const v8i*)(abase + 1024);
            v8i a2v = *(const v8i*)(abase + 2048);
            v8i a3v = *(const v8i*)(abase + 3072);
            acc0 = __builtin_amdgcn_wmma_i32_16x16x64_iu8(true, a0v, true, b, acc0, false, false);
            acc1 = __builtin_amdgcn_wmma_i32_16x16x64_iu8(true, a1v, true, b, acc1, false, false);
            acc2 = __builtin_amdgcn_wmma_i32_16x16x64_iu8(true, a2v, true, b, acc2, false, false);
            acc3 = __builtin_amdgcn_wmma_i32_16x16x64_iu8(true, a3v, true, b, acc3, false, false);
        }
    }
    #pragma unroll
    for (int r = 0; r < 8; ++r) {
        int co0 = ((moBase + 0) << 4) + half * 8 + r;
        int co1 = ((moBase + 1) << 4) + half * 8 + r;
        int co2 = ((moBase + 2) << 4) + half * 8 + r;
        int co3 = ((moBase + 3) << 4) + half * 8 + r;
        out[(size_t)co0 * PP + p] = scale[co0] * (float)acc0[r];
        out[(size_t)co1 * PP + p] = scale[co1] * (float)acc1[r];
        out[(size_t)co2 * PP + p] = scale[co2] * (float)acc2[r];
        out[(size_t)co3 * PP + p] = scale[co3] * (float)acc3[r];
    }
}

__global__ void k_gemm1(const signed char* __restrict__ sb,
                        const signed char* __restrict__ ap,
                        const float* __restrict__ scale,
                        float* __restrict__ out) {
    int wave = threadIdx.x >> 5;
    int lane = threadIdx.x & 31;
    int half = lane >> 4;
    int moBase = (wave & 3) * 4;
    int po = (blockIdx.x << 1) + (wave >> 2);
    int p  = (po << 4) + (lane & 15);

    const signed char* bp = sb + ((size_t)p << 8);
    v8i acc0 = {0,0,0,0,0,0,0,0};
    v8i acc1 = {0,0,0,0,0,0,0,0};
    v8i acc2 = {0,0,0,0,0,0,0,0};
    v8i acc3 = {0,0,0,0,0,0,0,0};
    #pragma unroll
    for (int ko = 0; ko < 4; ++ko) {
        const signed char* bb = bp + ko * 64 + half * 16;
        v4i b0 = *(const v4i*)bb;
        v4i b1 = *(const v4i*)(bb + 32);
        v8i b = cat8(b0, b1);
        const signed char* abase = ap + ((((size_t)ko) * 16 + moBase) << 10) + (lane << 5);
        v8i a0v = *(const v8i*)(abase);
        v8i a1v = *(const v8i*)(abase + 1024);
        v8i a2v = *(const v8i*)(abase + 2048);
        v8i a3v = *(const v8i*)(abase + 3072);
        acc0 = __builtin_amdgcn_wmma_i32_16x16x64_iu8(true, a0v, true, b, acc0, false, false);
        acc1 = __builtin_amdgcn_wmma_i32_16x16x64_iu8(true, a1v, true, b, acc1, false, false);
        acc2 = __builtin_amdgcn_wmma_i32_16x16x64_iu8(true, a2v, true, b, acc2, false, false);
        acc3 = __builtin_amdgcn_wmma_i32_16x16x64_iu8(true, a3v, true, b, acc3, false, false);
    }
    #pragma unroll
    for (int r = 0; r < 8; ++r) {
        int co0 = ((moBase + 0) << 4) + half * 8 + r;
        int co1 = ((moBase + 1) << 4) + half * 8 + r;
        int co2 = ((moBase + 2) << 4) + half * 8 + r;
        int co3 = ((moBase + 3) << 4) + half * 8 + r;
        out[(size_t)co0 * PP + p] = scale[co0] * (float)acc0[r];
        out[(size_t)co1 * PP + p] = scale[co1] * (float)acc1[r];
        out[(size_t)co2 * PP + p] = scale[co2] * (float)acc2[r];
        out[(size_t)co3 * PP + p] = scale[co3] * (float)acc3[r];
    }
}

__global__ void k_stats(const float* __restrict__ buf, const float* __restrict__ g,
                        const float* __restrict__ be, float* __restrict__ bnA,
                        float* __restrict__ bnB) {
    int c = blockIdx.x, t = threadIdx.x;
    const float* row = buf + (size_t)c * PP;
    float s = 0.f, ss = 0.f;
    for (int i = t; i < PP; i += 256) { float v = row[i]; s += v; ss += v * v; }
    __shared__ float sh[256], sh2[256];
    sh[t] = s; sh2[t] = ss;
    __syncthreads();
    for (int o = 128; o > 0; o >>= 1) {
        if (t < o) { sh[t] += sh[t + o]; sh2[t] += sh2[t + o]; }
        __syncthreads();
    }
    if (t == 0) {
        float mu  = sh[0] / (float)PP;
        float var = sh2[0] / (float)PP - mu * mu;
        float A = g[c] * rsqrtf(var + EPSV);
        bnA[c] = A;
        bnB[c] = be[c] - mu * A;
    }
}

__global__ void k_elem1(float* __restrict__ buf1, const float* __restrict__ x,
                        const float* __restrict__ b12, const float* __restrict__ b13,
                        const float* __restrict__ p1,  const float* __restrict__ b21,
                        const float* __restrict__ bnA, const float* __restrict__ bnB,
                        signed char* __restrict__ sb) {
    int c = blockIdx.y;
    int p = blockIdx.x * blockDim.x + threadIdx.x;
    size_t e = (size_t)c * PP + p;
    int n = p / HWS, hw = p - n * HWS;
    float xv = x[((size_t)n * CC + c) * HWS + hw];
    float t  = bnA[c] * buf1[e] + bnB[c];
    float u  = xv + t + b12[c];
    float o1 = (u >= 0.f ? u : p1[c] * u) + b13[c];
    buf1[e] = o1;
    sb[(size_t)p * CC + c] = sgn8(o1 + b21[c]);
}

__global__ void k_elem2(const float* __restrict__ buf2, const float* __restrict__ buf1,
                        const float* __restrict__ b22, const float* __restrict__ b23,
                        const float* __restrict__ p2,  const float* __restrict__ bnA,
                        const float* __restrict__ bnB, float* __restrict__ out) {
    int c = blockIdx.y;
    int p = blockIdx.x * blockDim.x + threadIdx.x;
    size_t e = (size_t)c * PP + p;
    float t = bnA[c] * buf2[e] + bnB[c] + buf1[e];
    float u = t + b22[c];
    float r = (u >= 0.f ? u : p2[c] * u) + b23[c];
    int n = p / HWS, hw = p - n * HWS;
    out[((size_t)n * CC + c) * HWS + hw] = r;
}

extern "C" void kernel_launch(void* const* d_in, const int* in_sizes, int n_in,
                              void* d_out, int out_size, void* d_ws, size_t ws_size,
                              hipStream_t stream) {
    const float* x    = (const float*)d_in[0];
    const float* b11  = (const float*)d_in[1];
    const float* b12  = (const float*)d_in[2];
    const float* b13  = (const float*)d_in[3];
    const float* b21  = (const float*)d_in[4];
    const float* b22  = (const float*)d_in[5];
    const float* b23  = (const float*)d_in[6];
    const float* w3   = (const float*)d_in[7];
    const float* wpw  = (const float*)d_in[8];
    const float* g1   = (const float*)d_in[9];
    const float* be1  = (const float*)d_in[10];
    const float* g2   = (const float*)d_in[11];
    const float* be2  = (const float*)d_in[12];
    const float* p1   = (const float*)d_in[13];
    const float* p2   = (const float*)d_in[14];
    float* out = (float*)d_out;

    char* ws = (char*)d_ws;
    size_t offSign = 0;
    size_t offBuf1 = offSign + (size_t)PP * CC;
    size_t offBuf2 = offBuf1 + (size_t)PP * CC * 4;
    size_t offAp1  = offBuf2 + (size_t)PP * CC * 4;
    size_t offAp2  = offAp1 + 9 * 4 * 16 * 1024;
    size_t offZ    = offAp2 + 4 * 16 * 1024;
    size_t offS1   = offZ + 1024;
    size_t offS2   = offS1 + 1024;
    size_t offA1   = offS2 + 1024;
    size_t offB1   = offA1 + 1024;
    size_t offA2   = offB1 + 1024;
    size_t offB2   = offA2 + 1024;

    signed char* sb   = (signed char*)(ws + offSign);
    float* buf1 = (float*)(ws + offBuf1);
    float* buf2 = (float*)(ws + offBuf2);
    signed char* ap1  = (signed char*)(ws + offAp1);
    signed char* ap2  = (signed char*)(ws + offAp2);
    signed char* zb   = (signed char*)(ws + offZ);
    float* s1   = (float*)(ws + offS1);
    float* s2   = (float*)(ws + offS2);
    float* bnA1 = (float*)(ws + offA1);
    float* bnB1 = (float*)(ws + offB1);
    float* bnA2 = (float*)(ws + offA2);
    float* bnB2 = (float*)(ws + offB2);

    k_scales<<<1, 256, 0, stream>>>(w3, wpw, s1, s2);
    k_zero<<<1, 64, 0, stream>>>((int*)zb);
    k_pack3<<<(9 * 4 * 16 * 1024) / 256, 256, 0, stream>>>(w3, ap1);
    k_pack1<<<(4 * 16 * 1024) / 256, 256, 0, stream>>>(wpw, ap2);

    k_sign1<<<(int)(((size_t)PP * CC) / 256), 256, 0, stream>>>(x, b11, sb);
    k_gemm3<<<PP / 32, 256, 0, stream>>>(sb, zb, ap1, s1, buf1);
    k_stats<<<256, 256, 0, stream>>>(buf1, g1, be1, bnA1, bnB1);
    k_elem1<<<dim3(PP / 256, 256), 256, 0, stream>>>(buf1, x, b12, b13, p1, b21,
                                                     bnA1, bnB1, sb);
    k_gemm1<<<PP / 32, 256, 0, stream>>>(sb, ap2, s2, buf2);
    k_stats<<<256, 256, 0, stream>>>(buf2, g2, be2, bnA2, bnB2);
    k_elem2<<<dim3(PP / 256, 256), 256, 0, stream>>>(buf2, buf1, b22, b23, p2,
                                                     bnA2, bnB2, out);
}
